// MarginLoss_16569983828456
// MI455X (gfx1250) — matrix operand order repaired and checked by reading, not compiled, not run on hardware
//
#include <hip/hip_runtime.h>
#include <math.h>

typedef _Float16 h8   __attribute__((ext_vector_type(8)));
typedef _Float16 v16h __attribute__((ext_vector_type(16)));
typedef float    v8f  __attribute__((ext_vector_type(8)));

#define N_ROWS   8192
#define N_DIM    256
#define MARGIN   0.5f
#define COLSPLIT 16                      // gridDim.y; 512 col tiles / 16 = 32 iters
#define NTILES   (N_ROWS / 16 / COLSPLIT)

// Monotone float <-> uint32 order-preserving encoding (for atomicMax on floats).
__device__ __forceinline__ unsigned f2ord(float f) {
    unsigned u = __float_as_uint(f);
    return (u & 0x80000000u) ? ~u : (u | 0x80000000u);
}
__device__ __forceinline__ float ord2f(unsigned o) {
    unsigned u = (o & 0x80000000u) ? (o ^ 0x80000000u) : ~o;
    return __uint_as_float(u);
}

// ---------------------------------------------------------------------------
// Kernel 1: L2-normalize rows (f32 in), emit f16 normalized embeddings.
// One wave32 per row; each lane owns 8 contiguous floats.
// ---------------------------------------------------------------------------
__global__ void __launch_bounds__(256)
normalize_rows(const float* __restrict__ emb, _Float16* __restrict__ ehat) {
    const int wave = threadIdx.x >> 5;
    const int lane = threadIdx.x & 31;
    const int row  = blockIdx.x * 8 + wave;

    const float* r = emb + (size_t)row * N_DIM + lane * 8;
    float4 x0 = ((const float4*)r)[0];
    float4 x1 = ((const float4*)r)[1];

    float s = x0.x*x0.x + x0.y*x0.y + x0.z*x0.z + x0.w*x0.w
            + x1.x*x1.x + x1.y*x1.y + x1.z*x1.z + x1.w*x1.w;
    #pragma unroll
    for (int off = 16; off >= 1; off >>= 1)
        s += __shfl_xor(s, off, 32);

    const float inv = 1.0f / fmaxf(sqrtf(s), 1e-12f);

    h8 o;
    o[0] = (_Float16)(x0.x * inv);
    o[1] = (_Float16)(x0.y * inv);
    o[2] = (_Float16)(x0.z * inv);
    o[3] = (_Float16)(x0.w * inv);
    o[4] = (_Float16)(x1.x * inv);
    o[5] = (_Float16)(x1.y * inv);
    o[6] = (_Float16)(x1.z * inv);
    o[7] = (_Float16)(x1.w * inv);
    *(h8*)(ehat + (size_t)row * N_DIM + lane * 8) = o;
}

// ---------------------------------------------------------------------------
// Kernel 2: fused  S = Ehat * Ehat^T  with masked per-row max epilogue.
//  - block = 4 waves; wave w owns A row-tile [i0, i0+16); full-K A fragment
//    (16x256 f16 = 64 VGPRs/lane) resident for the whole kernel.
//  - B tile (8KB) double-buffered in padded LDS: next tile's global loads
//    issue before the current tile's 8 WMMAs -> L2 latency hidden behind
//    matrix math; one workgroup barrier per iteration.
//  - Branchless masked-max epilogue (cndmask to -inf + v_max_num_f32),
//    no EXEC-mask branching in the hot loop.
// ---------------------------------------------------------------------------
__global__ void __launch_bounds__(128)
sim_maxes(const _Float16* __restrict__ ehat, const int* __restrict__ labels,
          unsigned* __restrict__ posbits, unsigned* __restrict__ negbits) {
    // Row stride 264 halfs = 132 dwords; 132 % 64 == 4 -> conflict-free b128.
    __shared__ __align__(16) _Float16 Bs[2][16][N_DIM + 8];

    const int lane = threadIdx.x & 31;
    const int wave = threadIdx.x >> 5;
    const int m    = lane & 15;   // row (A) / col (B) within tile
    const int h    = lane >> 4;   // half-wave selector
    const int i0   = blockIdx.x * 64 + wave * 16;

    // --- A fragments, ISA 16-bit A layout: lane m,h holds K = 16b + 8h + t ---
    v16h a[8];
    {
        const _Float16* arow = ehat + (size_t)(i0 + m) * N_DIM;
        #pragma unroll
        for (int s = 0; s < 8; ++s) {
            h8 c0 = *(const h8*)(arow + s * 32 + h * 8);
            h8 c1 = *(const h8*)(arow + s * 32 + 16 + h * 8);
            a[s] = __builtin_shufflevector(c0, c1,
                       0,1,2,3,4,5,6,7,8,9,10,11,12,13,14,15);
        }
    }

    // C-layout: acc element v <-> global row i0 + 8h + v. Preload row labels.
    int il[8];
    #pragma unroll
    for (int v = 0; v < 8; ++v) il[v] = labels[i0 + h * 8 + v];

    const float NEGINF = -__builtin_inff();
    float pos[8], neg[8];
    #pragma unroll
    for (int v = 0; v < 8; ++v) { pos[v] = NEGINF; neg[v] = NEGINF; }

    // cooperative B loader mapping: 128 threads, 32 halfs (64B) each
    const int ldr = threadIdx.x >> 3;         // B-tile row 0..15
    const int ldc = (threadIdx.x & 7) * 32;   // 32-half chunk

    // prime buffer 0 with the first tile
    h8 pf[4];
    {
        const _Float16* src = ehat + (size_t)(blockIdx.y * 16 + ldr) * N_DIM + ldc;
        #pragma unroll
        for (int q = 0; q < 4; ++q) pf[q] = *(const h8*)(src + q * 8);
        #pragma unroll
        for (int q = 0; q < 4; ++q) *(h8*)(&Bs[0][ldr][ldc + q * 8]) = pf[q];
    }

    for (int t = 0; t < NTILES; ++t) {
        const int jt  = blockIdx.y + t * COLSPLIT;
        const int j0  = jt * 16;
        const int buf = t & 1;

        __syncthreads();   // prior buffer stores visible; prior reads retired

        // issue next tile's global loads early (latency hidden by WMMAs)
        if (t + 1 < NTILES) {
            const _Float16* src =
                ehat + (size_t)((j0 + COLSPLIT * 16) + ldr) * N_DIM + ldc;
            #pragma unroll
            for (int q = 0; q < 4; ++q) pf[q] = *(const h8*)(src + q * 8);
        }

        v8f acc = {};
        #pragma unroll
        for (int s = 0; s < 8; ++s) {
            // ISA 16-bit B layout: lane n,h holds K = 16b + 8h + t (matches A)
            const _Float16* bp = &Bs[buf][m][s * 32 + h * 8];
            h8 b0 = *(const h8*)bp;
            h8 b1 = *(const h8*)(bp + 16);
            v16h b = __builtin_shufflevector(b0, b1,
                         0,1,2,3,4,5,6,7,8,9,10,11,12,13,14,15);
            acc = __builtin_amdgcn_wmma_f32_16x16x32_f16(
                      false, a[s], false, b, (short)0, acc, false, false);
        }

        // stash next tile into the other buffer (no conflict with reads above)
        if (t + 1 < NTILES) {
            #pragma unroll
            for (int q = 0; q < 4; ++q)
                *(h8*)(&Bs[buf ^ 1][ldr][ldc + q * 8]) = pf[q];
        }

        // branchless masked running maxes
        const int j  = j0 + m;
        const int jl = labels[j];
        #pragma unroll
        for (int v = 0; v < 8; ++v) {
            const int   i    = i0 + h * 8 + v;
            const float sv   = acc[v];
            const bool  same = (il[v] == jl);
            // diagonal always has same label -> only the positive max needs it
            const float sp = (same && (i != j)) ? sv : NEGINF;
            const float sn = same ? NEGINF : sv;
            pos[v] = fmaxf(pos[v], sp);
            neg[v] = fmaxf(neg[v], sn);
        }
    }

    // max across the 16 lanes holding each row (xor stays within half-wave)
    #pragma unroll
    for (int off = 1; off < 16; off <<= 1) {
        #pragma unroll
        for (int v = 0; v < 8; ++v) {
            pos[v] = fmaxf(pos[v], __shfl_xor(pos[v], off, 32));
            neg[v] = fmaxf(neg[v], __shfl_xor(neg[v], off, 32));
        }
    }

    if (m == 0) {
        #pragma unroll
        for (int v = 0; v < 8; ++v) {
            const int i = i0 + h * 8 + v;
            if (pos[v] != NEGINF) atomicMax(&posbits[i], f2ord(pos[v]));
            if (neg[v] != NEGINF) atomicMax(&negbits[i], f2ord(neg[v]));
        }
    }
}

// ---------------------------------------------------------------------------
// Kernel 3: loss = mean(relu(neg - pos + margin)); sentinel 0 => no valid
// candidate => 0.0 (matches reference's isfinite guard).
// ---------------------------------------------------------------------------
__global__ void __launch_bounds__(1024)
finalize_loss(const unsigned* __restrict__ posbits,
              const unsigned* __restrict__ negbits, float* __restrict__ out) {
    __shared__ float red[32];
    float local = 0.f;
    for (int i = threadIdx.x; i < N_ROWS; i += 1024) {
        const unsigned pb = posbits[i], nb = negbits[i];
        const float p = pb ? ord2f(pb) : 0.f;
        const float n = nb ? ord2f(nb) : 0.f;
        local += fmaxf(n - p + MARGIN, 0.f);
    }
    #pragma unroll
    for (int off = 16; off >= 1; off >>= 1) local += __shfl_xor(local, off, 32);
    if ((threadIdx.x & 31) == 0) red[threadIdx.x >> 5] = local;
    __syncthreads();
    if (threadIdx.x < 32) {
        float v = red[threadIdx.x];
        #pragma unroll
        for (int off = 16; off >= 1; off >>= 1) v += __shfl_xor(v, off, 32);
        if (threadIdx.x == 0) out[0] = v / (float)N_ROWS;
    }
}

// ---------------------------------------------------------------------------
extern "C" void kernel_launch(void* const* d_in, const int* in_sizes, int n_in,
                              void* d_out, int out_size, void* d_ws, size_t ws_size,
                              hipStream_t stream) {
    (void)in_sizes; (void)n_in; (void)out_size; (void)ws_size;
    const float* emb   = (const float*)d_in[0];
    const int*  labels = (const int*)d_in[1];
    float* out = (float*)d_out;

    char* ws = (char*)d_ws;
    _Float16* ehat    = (_Float16*)ws;                              // 4 MB
    unsigned* posbits = (unsigned*)(ws + (size_t)N_ROWS * N_DIM * 2);
    unsigned* negbits = posbits + N_ROWS;

    hipMemsetAsync(posbits, 0, 2 * N_ROWS * sizeof(unsigned), stream);
    normalize_rows<<<N_ROWS / 8, 256, 0, stream>>>(emb, ehat);
    sim_maxes<<<dim3(N_ROWS / 64, COLSPLIT), 128, 0, stream>>>(ehat, labels, posbits, negbits);
    finalize_loss<<<1, 1024, 0, stream>>>(posbits, negbits, out);
}
